// MultiScaleCAM_8366596292762
// MI455X (gfx1250) — matrix operand order repaired and checked by reading, not compiled, not run on hardware
//
#include <hip/hip_runtime.h>
#include <hip/hip_bf16.h>

// ---------------------------------------------------------------------------
// MultiScaleCAM for MI455X (gfx1250, wave32)
//   Phase A: bucketed CAM reads (fine=4 slots, coarse=32 slots), wave/token,
//            L2-resident gathers, wave32 butterfly reductions.
//   Phase B: W_gate [1024,512] f32 -> Wt [512,1024] bf16 (transposed).
//   Phase C: gate GEMM [ntok,1024]x[1024,512] via v_wmma_f32_16x16x32_bf16,
//            16x64 tile per wave (A-fragment reuse x4),
//            fused bias + sigmoid + gated blend epilogue.
// ---------------------------------------------------------------------------

typedef __attribute__((ext_vector_type(16))) __bf16 v16bf;
typedef __attribute__((ext_vector_type(8)))  float  v8f;

#define DMODEL 512
#define SPB_F  4
#define SPB_C  32
#define NB_F   4096
#define NB_C   512

__device__ __forceinline__ float waveReduceSum(float v) {
    // wave32 butterfly: all lanes end with the full sum
    #pragma unroll
    for (int off = 16; off > 0; off >>= 1)
        v += __shfl_xor(v, off, 32);
    return v;
}

__device__ __forceinline__ float dot4(const float4 a, const float4 b) {
    return fmaf(a.x, b.x, fmaf(a.y, b.y, fmaf(a.z, b.z, a.w * b.w)));
}

__device__ __forceinline__ unsigned short f2bf(float f) {
    unsigned u = __builtin_bit_cast(unsigned, f);
    u += 0x7FFFu + ((u >> 16) & 1u);      // round-to-nearest-even
    return (unsigned short)(u >> 16);
}

__device__ __forceinline__ unsigned pack2(float a, float b) {
    return (unsigned)f2bf(a) | ((unsigned)f2bf(b) << 16);
}

// ---------------------------------------------------------------------------
// Phase A: one wave per token. Lane owns 16 consecutive dims (4 x float4).
// ---------------------------------------------------------------------------
__global__ __launch_bounds__(256) void cam_read_kernel(
    const float* __restrict__ query, const int* __restrict__ tids,
    const float* __restrict__ svf, const float* __restrict__ skf,
    const float* __restrict__ svc, const float* __restrict__ skc,
    float* __restrict__ vf, float* __restrict__ vc,
    unsigned short* __restrict__ gin, int ntok)
{
    const int wave = threadIdx.x >> 5;
    const int lane = threadIdx.x & 31;
    const int t = blockIdx.x * 8 + wave;
    if (t >= ntok) return;                       // wave-uniform exit

    const int dofs = lane * 16;
    const float4* qp = (const float4*)(query + (size_t)t * DMODEL + dofs);
    float4 q0 = qp[0], q1 = qp[1], q2 = qp[2], q3 = qp[3];

    // 1/(||q|| + eps): fold the normalization into the sims
    float ss = dot4(q0, q0) + dot4(q1, q1) + dot4(q2, q2) + dot4(q3, q3);
    ss = waveReduceSum(ss);
    const float rnorm = 1.0f / (sqrtf(ss) + 1e-8f);

    const int tid = tids[t];
    const size_t fb = (size_t)(tid & (NB_F - 1)) * SPB_F;   // fine slot base
    const size_t cb = (size_t)(tid & (NB_C - 1)) * SPB_C;   // coarse slot base

    // ---------------- fine: 4 slots ----------------
    float simf[SPB_F];
    #pragma unroll
    for (int p = 0; p < SPB_F; ++p) {
        const float4* kp = (const float4*)(skf + (fb + p) * DMODEL + dofs);
        __builtin_prefetch(svf + (fb + p) * DMODEL + dofs, 0, 0);  // global_prefetch_b8
        float s = dot4(q0, kp[0]) + dot4(q1, kp[1]) + dot4(q2, kp[2]) + dot4(q3, kp[3]);
        simf[p] = waveReduceSum(s) * rnorm;
    }
    float mx = simf[0];
    #pragma unroll
    for (int p = 1; p < SPB_F; ++p) mx = fmaxf(mx, simf[p]);
    float se = 0.0f;
    #pragma unroll
    for (int p = 0; p < SPB_F; ++p) { simf[p] = __expf(simf[p] - mx); se += simf[p]; }
    const float invf = 1.0f / se;

    float4 a0 = {0,0,0,0}, a1 = {0,0,0,0}, a2 = {0,0,0,0}, a3 = {0,0,0,0};
    #pragma unroll
    for (int p = 0; p < SPB_F; ++p) {
        const float w = simf[p] * invf;
        const float4* vp = (const float4*)(svf + (fb + p) * DMODEL + dofs);
        float4 x;
        x = vp[0]; a0.x = fmaf(w, x.x, a0.x); a0.y = fmaf(w, x.y, a0.y); a0.z = fmaf(w, x.z, a0.z); a0.w = fmaf(w, x.w, a0.w);
        x = vp[1]; a1.x = fmaf(w, x.x, a1.x); a1.y = fmaf(w, x.y, a1.y); a1.z = fmaf(w, x.z, a1.z); a1.w = fmaf(w, x.w, a1.w);
        x = vp[2]; a2.x = fmaf(w, x.x, a2.x); a2.y = fmaf(w, x.y, a2.y); a2.z = fmaf(w, x.z, a2.z); a2.w = fmaf(w, x.w, a2.w);
        x = vp[3]; a3.x = fmaf(w, x.x, a3.x); a3.y = fmaf(w, x.y, a3.y); a3.z = fmaf(w, x.z, a3.z); a3.w = fmaf(w, x.w, a3.w);
    }
    // alpha = 0.5 residual blend
    float4 f0, f1, f2, f3;
    f0.x = 0.5f*a0.x + 0.5f*q0.x; f0.y = 0.5f*a0.y + 0.5f*q0.y; f0.z = 0.5f*a0.z + 0.5f*q0.z; f0.w = 0.5f*a0.w + 0.5f*q0.w;
    f1.x = 0.5f*a1.x + 0.5f*q1.x; f1.y = 0.5f*a1.y + 0.5f*q1.y; f1.z = 0.5f*a1.z + 0.5f*q1.z; f1.w = 0.5f*a1.w + 0.5f*q1.w;
    f2.x = 0.5f*a2.x + 0.5f*q2.x; f2.y = 0.5f*a2.y + 0.5f*q2.y; f2.z = 0.5f*a2.z + 0.5f*q2.z; f2.w = 0.5f*a2.w + 0.5f*q2.w;
    f3.x = 0.5f*a3.x + 0.5f*q3.x; f3.y = 0.5f*a3.y + 0.5f*q3.y; f3.z = 0.5f*a3.z + 0.5f*q3.z; f3.w = 0.5f*a3.w + 0.5f*q3.w;

    float4* vfo = (float4*)(vf + (size_t)t * DMODEL + dofs);
    vfo[0] = f0; vfo[1] = f1; vfo[2] = f2; vfo[3] = f3;

    {   // bf16 copy into g_in[:, 0:512]
        uint4 g0 = { pack2(f0.x,f0.y), pack2(f0.z,f0.w), pack2(f1.x,f1.y), pack2(f1.z,f1.w) };
        uint4 g1 = { pack2(f2.x,f2.y), pack2(f2.z,f2.w), pack2(f3.x,f3.y), pack2(f3.z,f3.w) };
        unsigned short* gp = gin + (size_t)t * (2*DMODEL) + dofs;
        *(uint4*)(gp)     = g0;
        *(uint4*)(gp + 8) = g1;
    }

    // ---------------- coarse: 32 slots ----------------
    float sc[SPB_C];
    #pragma unroll 4
    for (int p = 0; p < SPB_C; ++p) {
        const float4* kp = (const float4*)(skc + (cb + p) * DMODEL + dofs);
        __builtin_prefetch(svc + (cb + p) * DMODEL + dofs, 0, 0);
        float s = dot4(q0, kp[0]) + dot4(q1, kp[1]) + dot4(q2, kp[2]) + dot4(q3, kp[3]);
        sc[p] = waveReduceSum(s) * rnorm;
    }
    mx = sc[0];
    #pragma unroll
    for (int p = 1; p < SPB_C; ++p) mx = fmaxf(mx, sc[p]);
    se = 0.0f;
    #pragma unroll
    for (int p = 0; p < SPB_C; ++p) { sc[p] = __expf(sc[p] - mx); se += sc[p]; }
    const float invc = 1.0f / se;

    a0 = {0,0,0,0}; a1 = {0,0,0,0}; a2 = {0,0,0,0}; a3 = {0,0,0,0};
    #pragma unroll 4
    for (int p = 0; p < SPB_C; ++p) {
        const float w = sc[p] * invc;
        const float4* vp = (const float4*)(svc + (cb + p) * DMODEL + dofs);
        float4 x;
        x = vp[0]; a0.x = fmaf(w, x.x, a0.x); a0.y = fmaf(w, x.y, a0.y); a0.z = fmaf(w, x.z, a0.z); a0.w = fmaf(w, x.w, a0.w);
        x = vp[1]; a1.x = fmaf(w, x.x, a1.x); a1.y = fmaf(w, x.y, a1.y); a1.z = fmaf(w, x.z, a1.z); a1.w = fmaf(w, x.w, a1.w);
        x = vp[2]; a2.x = fmaf(w, x.x, a2.x); a2.y = fmaf(w, x.y, a2.y); a2.z = fmaf(w, x.z, a2.z); a2.w = fmaf(w, x.w, a2.w);
        x = vp[3]; a3.x = fmaf(w, x.x, a3.x); a3.y = fmaf(w, x.y, a3.y); a3.z = fmaf(w, x.z, a3.z); a3.w = fmaf(w, x.w, a3.w);
    }
    // alpha = 1.0: v_coarse = v_mem
    float4* vco = (float4*)(vc + (size_t)t * DMODEL + dofs);
    vco[0] = a0; vco[1] = a1; vco[2] = a2; vco[3] = a3;

    {   // bf16 copy into g_in[:, 512:1024]
        uint4 g0 = { pack2(a0.x,a0.y), pack2(a0.z,a0.w), pack2(a1.x,a1.y), pack2(a1.z,a1.w) };
        uint4 g1 = { pack2(a2.x,a2.y), pack2(a2.z,a2.w), pack2(a3.x,a3.y), pack2(a3.z,a3.w) };
        unsigned short* gp = gin + (size_t)t * (2*DMODEL) + DMODEL + dofs;
        *(uint4*)(gp)     = g0;
        *(uint4*)(gp + 8) = g1;
    }
}

// ---------------------------------------------------------------------------
// Phase B: W_gate [1024,512] f32 row-major  ->  Wt [512,1024] bf16 (N-major)
// ---------------------------------------------------------------------------
__global__ __launch_bounds__(256) void wconvert_kernel(
    const float* __restrict__ W, unsigned short* __restrict__ Wt)
{
    const int idx = blockIdx.x * 256 + threadIdx.x;      // 0 .. 1024*512-1
    const int k = idx >> 9;                              // row of W   (K)
    const int n = idx & 511;                             // col of W   (N)
    Wt[(size_t)n * 1024 + k] = f2bf(W[idx]);             // coalesced read
}

// ---------------------------------------------------------------------------
// Phase C: one wave per 16x64 tile (4 accumulators, A reused x4 per K-step).
//   A (16x32 bf16): lane l -> row M=l&15; K halves select by l>=16.
//   B (32x16 bf16): lane l -> col N=l&15; K = 8h..8h+7 and 16+8h..+7 per half.
//   C (16x16 f32):  VGPR r, lane l -> M = r + 8*(l>=16), N = l&15.
// ---------------------------------------------------------------------------
__global__ __launch_bounds__(256) void gate_gemm_kernel(
    const unsigned short* __restrict__ gin, const unsigned short* __restrict__ Wt,
    const float* __restrict__ bias,
    const float* __restrict__ vf, const float* __restrict__ vc,
    float* __restrict__ out, int ntok)
{
    const int wave = threadIdx.x >> 5;
    const int lane = threadIdx.x & 31;
    const int gid  = blockIdx.x * 8 + wave;
    const int tilesN = DMODEL / 64;                      // 8 tiles of 64 cols
    const int tilesM = ntok / 16;
    if (gid >= tilesM * tilesN) return;                  // wave-uniform exit

    const int mtile = (gid / tilesN) * 16;
    const int nbase = (gid % tilesN) * 64;
    const int half  = lane >> 4;                         // 0: K-lo, 1: K-hi
    const int lr    = lane & 15;

    const unsigned short* arow = gin + (size_t)(mtile + lr) * (2*DMODEL);
    const unsigned short* brow[4];
    #pragma unroll
    for (int j = 0; j < 4; ++j)
        brow[j] = Wt + (size_t)(nbase + j * 16 + lr) * (2*DMODEL);

    union Frag { uint4 u[2]; v16bf v; };
    v8f c[4] = {{}, {}, {}, {}};

    #pragma unroll 2
    for (int ko = 0; ko < 2*DMODEL; ko += 32) {
        Frag A;
        // A row M: K = ko+8h .. +7  and  ko+16+8h .. +7   (8 bf16 = b128 each)
        A.u[0] = *(const uint4*)(arow + ko + half * 8);
        A.u[1] = *(const uint4*)(arow + ko + 16 + half * 8);
        #pragma unroll
        for (int j = 0; j < 4; ++j) {
            Frag B;
            // B col N: K = ko+8h .. +7  and  ko+16+8h .. +7   (8 bf16 = b128 each)
            B.u[0] = *(const uint4*)(brow[j] + ko + half * 8);
            B.u[1] = *(const uint4*)(brow[j] + ko + 16 + half * 8);
            c[j] = __builtin_amdgcn_wmma_f32_16x16x32_bf16(
                    /*neg_a=*/false, A.v, /*neg_b=*/false, B.v,
                    /*c_mod=*/(short)0, c[j], /*reuse_a=*/false, /*reuse_b=*/false);
        }
    }

    #pragma unroll
    for (int j = 0; j < 4; ++j) {
        const int ncol = nbase + j * 16 + lr;
        const float bn = bias[ncol];
        #pragma unroll
        for (int r = 0; r < 8; ++r) {
            const int m = mtile + r + half * 8;
            const size_t idx = (size_t)m * DMODEL + ncol;
            const float g = 1.0f / (1.0f + __expf(-(c[j][r] + bn)));
            out[idx] = (1.0f - g) * vc[idx] + g * vf[idx];
        }
    }
}

// ---------------------------------------------------------------------------
extern "C" void kernel_launch(void* const* d_in, const int* in_sizes, int n_in,
                              void* d_out, int out_size, void* d_ws, size_t ws_size,
                              hipStream_t stream) {
    const float* query = (const float*)d_in[0];
    const int*   tids  = (const int*)  d_in[1];
    const float* svf   = (const float*)d_in[2];
    const float* skf   = (const float*)d_in[3];
    const float* svc   = (const float*)d_in[4];
    const float* skc   = (const float*)d_in[5];
    const float* W     = (const float*)d_in[6];
    const float* bgate = (const float*)d_in[7];
    float* out = (float*)d_out;

    const int ntok = in_sizes[1];                        // B*S = 8192

    // workspace carve-up
    float* vf = (float*)d_ws;                            // ntok*512 f32
    float* vc = vf + (size_t)ntok * DMODEL;              // ntok*512 f32
    unsigned short* gin = (unsigned short*)(vc + (size_t)ntok * DMODEL); // ntok*1024 bf16
    unsigned short* wt  = gin + (size_t)ntok * 2 * DMODEL;               // 512*1024 bf16

    // Phase A: CAM reads (wave per token, 8 waves/block)
    cam_read_kernel<<<(ntok + 7) / 8, 256, 0, stream>>>(
        query, tids, svf, skf, svc, skc, vf, vc, gin, ntok);

    // Phase B: W_gate -> bf16 transposed
    wconvert_kernel<<<(1024 * 512) / 256, 256, 0, stream>>>(W, wt);

    // Phase C: WMMA gate GEMM (16x64 per wave) + fused sigmoid blend
    const int tiles = (ntok / 16) * (DMODEL / 64);
    gate_gemm_kernel<<<(tiles + 7) / 8, 256, 0, stream>>>(
        gin, wt, bgate, vf, vc, out, ntok);
}
